// SpatialHRVQTokenizer_91225105367464
// MI455X (gfx1250) — compile-verified, matrix operand order checked by reading, not run on hardware
//
#include <hip/hip_runtime.h>

typedef __attribute__((ext_vector_type(16))) __bf16         v16bf;
typedef __attribute__((ext_vector_type(8)))  float          v8f;
typedef __attribute__((ext_vector_type(16))) unsigned short u16x16;
typedef __attribute__((ext_vector_type(8)))  unsigned short u16x8;

#define D_MODEL 384
#define KCH     12      // 384 / 32 K-chunks
#define NCODES  256
#define NTILES  16      // 256 / 16 code tiles
#define SB      392     // padded LDS row stride in shorts (784B -> bank base 4*lane, conflict-free)
#define WAVES   8       // 256 threads, 128 rows per workgroup

__device__ __forceinline__ unsigned short bf16_rne(float f) {
    unsigned u = __float_as_uint(f);
    unsigned r = 0x7FFFu + ((u >> 16) & 1u);
    return (unsigned short)((u + r) >> 16);
}

// split f32 into bf16 hi + bf16 lo so that hi + lo ~= f (near-f32 GEMM accuracy)
__device__ __forceinline__ void split2bf16(float f, unsigned short& h, unsigned short& l) {
    unsigned u  = __float_as_uint(f);
    unsigned uh = u + (0x7FFFu + ((u >> 16) & 1u));
    h = (unsigned short)(uh >> 16);
    float fh = __uint_as_float(uh & 0xFFFF0000u);
    l = bf16_rne(f - fh);
}

__device__ __forceinline__ v8f wmma_bf16f32(v16bf a, v16bf b, v8f c) {
    return __builtin_amdgcn_wmma_f32_16x16x32_bf16(false, a, false, b, (short)0, c, false, false);
}

__device__ __forceinline__ v16bf make_b16x16(u16x8 lo8, u16x8 hi8) {
    u16x16 t = {};
#pragma unroll
    for (int e = 0; e < 8; ++e) { t[e] = lo8[e]; t[8 + e] = hi8[e]; }
    return __builtin_bit_cast(v16bf, t);
}

// ---------------- prep kernels ----------------
__global__ void prep_convert(const float* __restrict__ c0, const float* __restrict__ c1,
                             const float* __restrict__ c2,
                             unsigned short* __restrict__ hi, unsigned short* __restrict__ lo) {
    int i = blockIdx.x * blockDim.x + threadIdx.x;
    const int per = NCODES * D_MODEL;
    if (i >= 3 * per) return;
    int lvl = i / per;
    int e   = i - lvl * per;
    const float* src = (lvl == 0) ? c0 : (lvl == 1) ? c1 : c2;
    unsigned short h, l;
    split2bf16(src[e], h, l);
    hi[i] = h; lo[i] = l;
}

__global__ void prep_norms(const float* __restrict__ c0, const float* __restrict__ c1,
                           const float* __restrict__ c2,
                           float* __restrict__ norms, float* __restrict__ loss_acc) {
    int lvl = blockIdx.x;           // 3 blocks
    int c   = threadIdx.x;          // 256 codes
    const float* src = (lvl == 0) ? c0 : (lvl == 1) ? c1 : c2;
    float s = 0.0f;
    for (int k = 0; k < D_MODEL; ++k) { float f = src[c * D_MODEL + k]; s += f * f; }
    norms[lvl * NCODES + c] = s;
    if (c == 0) loss_acc[lvl] = 0.0f;   // re-zero every launch (graph-replay safe)
}

// ---------------- main VQ kernel ----------------
// min 4 waves/SIMD: forces the VGPR budget to <=256 so the allocator stays out
// of s_set_vgpr_msb territory (loop working set is ~190 VGPRs).
__global__ __launch_bounds__(256, 4)
void vq_level_kernel(const float* __restrict__ x,
                     const float* __restrict__ cbf,            // (256,384) f32 original
                     const unsigned short* __restrict__ cbhi,  // (256,384) bf16 hi
                     const unsigned short* __restrict__ cblo,  // (256,384) bf16 lo
                     const float* __restrict__ norms,          // (256) ||e||^2
                     float* __restrict__ out_idx,              // (R) idx as float
                     float* __restrict__ out_q,                // (R,384)
                     float* __restrict__ loss_acc,             // scalar accumulator
                     int R) {
    (void)R;
    __shared__ __align__(16) unsigned short s_bhi[16 * SB];          // 12.5 KB
    __shared__ __align__(16) unsigned short s_blo[16 * SB];          // 12.5 KB
    __shared__ __align__(16) unsigned short s_alo[WAVES * 16 * SB];  // 100 KB (per-wave A_lo)
    __shared__ float s_norms[NCODES];
    __shared__ float s_rowval[WAVES * 16];
    __shared__ int   s_rowidx[WAVES * 16];

    const int tid  = threadIdx.x;
    const int wv   = tid >> 5;
    const int lane = tid & 31;
    const int half = lane >> 4;
    const int l15  = lane & 15;

    const int rowbase = blockIdx.x * (WAVES * 16) + wv * 16;
    const int row     = rowbase + l15;

    s_norms[tid & (NCODES - 1)] = norms[tid & (NCODES - 1)];

    // ---- load A tile (16 rows x 384): bf16-hi resident in VGPRs, bf16-lo to LDS.
    // 16-bit A layout: lane l holds row (l&15); elems 0-7 at K = c*32 + half*8,
    // elems 8-15 at K = c*32 + 16 + half*8.
    v16bf a_hi[KCH];
    unsigned short* aw = s_alo + wv * 16 * SB;
    const int abase = l15 * SB + half * 8;
    float xnp = 0.0f;
    {
        const float* px = x + (size_t)row * D_MODEL + half * 8;
#pragma unroll
        for (int c = 0; c < KCH; ++c) {
            u16x16 th = {};
            u16x8  tl0 = {}, tl1 = {};
            // per-float4 convert keeps each load's live range at 4 floats so the
            // backend can keep the wide b128 loads under a tight VGPR budget
#pragma unroll
            for (int g = 0; g < 4; ++g) {
                const int koff = (g < 2) ? (g * 4) : (8 + g * 4);   // 0,4,16,20
                float4 f = *(const float4*)(px + c * 32 + koff);
                float vv[4] = {f.x, f.y, f.z, f.w};
#pragma unroll
                for (int e = 0; e < 4; ++e) {
                    xnp += vv[e] * vv[e];
                    unsigned short h, l;
                    split2bf16(vv[e], h, l);
                    int idx = g * 4 + e;        // 0..15 in v16 element order
                    th[idx] = h;
                    if (idx < 8) tl0[idx] = l; else tl1[idx - 8] = l;
                }
            }
            a_hi[c] = __builtin_bit_cast(v16bf, th);
            *(u16x8*)&aw[abase + c * 32]      = tl0;
            *(u16x8*)&aw[abase + c * 32 + 16] = tl1;
            // Cap live ranges: one chunk (4 global loads) in flight at a time.
            __builtin_amdgcn_sched_barrier(0);
        }
    }
    // lanes l and l+16 together cover all K of row (l&15)
    float xnorm = xnp + __shfl_xor(xnp, 16, 32);

    float minv[8];
    int   mini[8];
#pragma unroll
    for (int j = 0; j < 8; ++j) { minv[j] = 3.0e38f; mini[j] = 0; }

    const int bbase = l15 * SB + half * 8;   // B layout mirrors A: lane = code column

#pragma unroll 1
    for (int t = 0; t < NTILES; ++t) {
        __syncthreads();
        // stage 16 codes x 384 (hi+lo) into LDS with padded rows (48 uint4 -> stride 49)
        {
            const uint4* ghi = (const uint4*)(cbhi + t * 16 * D_MODEL);
            const uint4* glo = (const uint4*)(cblo + t * 16 * D_MODEL);
            uint4* shi = (uint4*)s_bhi;
            uint4* slo = (uint4*)s_blo;
            for (int i = tid; i < 768; i += 256) {
                int r = i / 48, cc = i - r * 48;
                shi[r * 49 + cc] = ghi[i];
                slo[r * 49 + cc] = glo[i];
            }
        }
        __syncthreads();

        v8f acc_hh = {0.f, 0.f, 0.f, 0.f, 0.f, 0.f, 0.f, 0.f};
        v8f acc_hl = {0.f, 0.f, 0.f, 0.f, 0.f, 0.f, 0.f, 0.f};
        v8f acc_lh = {0.f, 0.f, 0.f, 0.f, 0.f, 0.f, 0.f, 0.f};
#pragma unroll
        for (int c = 0; c < KCH; ++c) {
            u16x8 bh0 = *(const u16x8*)&s_bhi[bbase + c * 32];
            u16x8 bh1 = *(const u16x8*)&s_bhi[bbase + c * 32 + 16];
            u16x8 bl0 = *(const u16x8*)&s_blo[bbase + c * 32];
            u16x8 bl1 = *(const u16x8*)&s_blo[bbase + c * 32 + 16];
            u16x8 al0 = *(const u16x8*)&aw[abase + c * 32];
            u16x8 al1 = *(const u16x8*)&aw[abase + c * 32 + 16];
            v16bf bh = make_b16x16(bh0, bh1);
            v16bf bl = make_b16x16(bl0, bl1);
            v16bf al = make_b16x16(al0, al1);
            acc_hh = wmma_bf16f32(a_hi[c], bh, acc_hh);
            acc_hl = wmma_bf16f32(a_hi[c], bl, acc_hl);
            acc_lh = wmma_bf16f32(al,      bh, acc_lh);
        }
        const int   code = t * 16 + l15;
        const float nrm  = s_norms[code];
#pragma unroll
        for (int j = 0; j < 8; ++j) {
            float d = nrm - 2.0f * ((acc_hh[j] + acc_hl[j]) + acc_lh[j]);  // ||e||^2 - 2 x.e
            if (d < minv[j]) { minv[j] = d; mini[j] = code; }
        }
    }

    // butterfly arg-min across the 16 columns (halves stay separate: masks < 16)
#pragma unroll
    for (int m = 1; m <= 8; m <<= 1) {
#pragma unroll
        for (int j = 0; j < 8; ++j) {
            float ov = __shfl_xor(minv[j], m, 32);
            int   oi = __shfl_xor(mini[j], m, 32);
            if (ov < minv[j] || (ov == minv[j] && oi < mini[j])) { minv[j] = ov; mini[j] = oi; }
        }
    }
    // lanes 0-15 hold rows 0-7 (vgpr j), lanes 16-31 hold rows 8-15; publish per-wave
    if (lane == 0) {
#pragma unroll
        for (int j = 0; j < 8; ++j) { s_rowval[wv * 16 + j] = minv[j]; s_rowidx[wv * 16 + j] = mini[j]; }
    }
    if (lane == 16) {
#pragma unroll
        for (int j = 0; j < 8; ++j) { s_rowval[wv * 16 + 8 + j] = minv[j]; s_rowidx[wv * 16 + 8 + j] = mini[j]; }
    }
    // same-wave DS ops are in-order: safe to read back without a block barrier
    float rv = s_rowval[wv * 16 + l15];
    int   ri = s_rowidx[wv * 16 + l15];

    // loss: ||q-x||^2 per row = d2_min + ||x||^2 ; lane r (r<16) owns row r
    float sse = (lane < 16) ? (rv + xnorm) : 0.0f;
    if (lane < 16) out_idx[rowbase + l15] = (float)ri;
#pragma unroll
    for (int m = 1; m <= 16; m <<= 1) sse += __shfl_xor(sse, m, 32);
    if (lane == 0) atomicAdd(loss_acc, sse);

    // gather q rows from the exact f32 codebook: 96 float4 per row, whole wave
#pragma unroll 1
    for (int r = 0; r < 16; ++r) {
        int code = s_rowidx[wv * 16 + r];
        const float4* src = (const float4*)(cbf + (size_t)code * D_MODEL);
        float4*       dst = (float4*)(out_q + (size_t)(rowbase + r) * D_MODEL);
#pragma unroll
        for (int it = 0; it < 3; ++it) dst[it * 32 + lane] = src[it * 32 + lane];
    }
}

__global__ void finalize_loss(const float* __restrict__ loss_acc, float* __restrict__ out) {
    if (threadIdx.x == 0 && blockIdx.x == 0) {
        float t = 0.05f * loss_acc[0] / (32768.0f * 384.0f)
                + 0.25f * loss_acc[1] / (131072.0f * 384.0f)
                + 0.60f * loss_acc[2] / (131072.0f * 384.0f);
        *out = t;
    }
}

extern "C" void kernel_launch(void* const* d_in, const int* in_sizes, int n_in,
                              void* d_out, int out_size, void* d_ws, size_t ws_size,
                              hipStream_t stream) {
    (void)in_sizes; (void)n_in; (void)out_size; (void)ws_size;
    const float* l0 = (const float*)d_in[0];
    const float* l1 = (const float*)d_in[1];
    const float* l2 = (const float*)d_in[2];
    const float* c0 = (const float*)d_in[3];
    const float* c1 = (const float*)d_in[4];
    const float* c2 = (const float*)d_in[5];
    float* out = (float*)d_out;

    const int R0 = 8192 * 4, R1 = 8192 * 16, R2 = 8192 * 16;
    size_t off = 0;
    float* o_idx0 = out + off; off += (size_t)R0;
    float* o_idx1 = out + off; off += (size_t)R1;
    float* o_idx2 = out + off; off += (size_t)R2;
    float* o_loss = out + off; off += 1;
    float* o_q0   = out + off; off += (size_t)R0 * D_MODEL;
    float* o_q1   = out + off; off += (size_t)R1 * D_MODEL;
    float* o_q2   = out + off;

    const int per = NCODES * D_MODEL;               // 98304 elems per codebook
    unsigned short* hi = (unsigned short*)d_ws;     // 3*per shorts
    unsigned short* lo = hi + 3 * per;              // 3*per shorts
    float* norms = (float*)(lo + 3 * per);          // 3*256 floats
    float* lacc  = norms + 3 * NCODES;              // 3 floats

    prep_convert<<<(3 * per + 255) / 256, 256, 0, stream>>>(c0, c1, c2, hi, lo);
    prep_norms<<<3, 256, 0, stream>>>(c0, c1, c2, norms, lacc);

    const int RPB = WAVES * 16;  // 128 rows per block
    vq_level_kernel<<<R0 / RPB, 256, 0, stream>>>(l0, c0, hi,           lo,           norms,       o_idx0, o_q0, lacc + 0, R0);
    vq_level_kernel<<<R1 / RPB, 256, 0, stream>>>(l1, c1, hi + per,     lo + per,     norms + 256, o_idx1, o_q1, lacc + 1, R1);
    vq_level_kernel<<<R2 / RPB, 256, 0, stream>>>(l2, c2, hi + 2 * per, lo + 2 * per, norms + 512, o_idx2, o_q2, lacc + 2, R2);

    finalize_loss<<<1, 1, 0, stream>>>(lacc, o_loss);
}
